// OneToManyRNN_6399501271108
// MI455X (gfx1250) — compile-verified, matrix operand order checked by reading, not run on hardware
//
#include <hip/hip_runtime.h>
#include <hip/hip_bf16.h>

// ---------------------------------------------------------------------------
// OneToManyRNN (LSTM + Dense per step), B=64, T=2048, D=H=O=256.
//   K1: gates_pre = x @ Wih + bh        (parallel bf16 WMMA GEMM, whole chip)
//   K2: recurrence only: h_t stream     (4 WGs, 16 batch rows each; Whh gates
//       i,f resident in LDS (256KB), gates g,o streamed from L2 w/ double
//       buffer; gates_pre prefetched one step ahead; 1 barrier/step)
//   K3: out = hseq @ Wout + bout        (parallel bf16 WMMA GEMM, whole chip)
// ---------------------------------------------------------------------------

#define B_  64
#define T_  2048
#define D_  256
#define H_  256
#define O_  256
#define G4H 1024
#define BT  (B_ * T_)          // 131072

typedef __bf16 v16bf __attribute__((ext_vector_type(16)));
typedef __bf16 v8bf  __attribute__((ext_vector_type(8)));
typedef float  v8f   __attribute__((ext_vector_type(8)));

// workspace layout (bytes)
#define GP_OFF     0u                                  // gates_pre bf16 [BT][1024]
#define GP_BYTES   ((size_t)BT * G4H * 2)
#define XB_OFF     (GP_OFF + GP_BYTES)                 // x bf16 [BT][256]
#define XB_BYTES   ((size_t)BT * D_ * 2)
#define HS_OFF     (XB_OFF + XB_BYTES)                 // hseq bf16 [B][T][256]
#define HS_BYTES   ((size_t)BT * H_ * 2)
#define WIHT_OFF   (HS_OFF + HS_BYTES)                 // WihT bf16 [1024][256]
#define WIHT_BYTES ((size_t)G4H * D_ * 2)
#define WHHT_OFF   (WIHT_OFF + WIHT_BYTES)             // WhhT bf16 [1024][256]
#define WHHT_BYTES ((size_t)G4H * H_ * 2)
#define WOUTT_OFF  (WHHT_OFF + WHHT_BYTES)             // WoutT bf16 [256][256]

__device__ __forceinline__ v16bf combine8(v8bf lo, v8bf hi) {
  return __builtin_shufflevector(lo, hi, 0,1,2,3,4,5,6,7,8,9,10,11,12,13,14,15);
}
__device__ __forceinline__ v16bf load_frag(const __bf16* p) {
  return combine8(*(const v8bf*)p, *(const v8bf*)(p + 16));
}
__device__ __forceinline__ v8f wmma_bf16(v16bf a, v16bf b, v8f c) {
  return __builtin_amdgcn_wmma_f32_16x16x32_bf16(false, a, false, b,
                                                 (short)0, c, false, false);
}
__device__ __forceinline__ float sigf(float x) {
  return 1.0f / (1.0f + __expf(-x));
}
__device__ __forceinline__ float tanh_fast(float x) {
  float e = __expf(-2.0f * x);
  return (1.0f - e) / (1.0f + e);
}

// --------------------------- conversion kernels ----------------------------

__global__ __launch_bounds__(256) void cvt_f32_to_bf16(const float* __restrict__ in,
                                                       __bf16* __restrict__ out, int n) {
  int i = blockIdx.x * 256 + threadIdx.x;
  if (i < n) out[i] = (__bf16)in[i];
}

// in[rows][cols] (f32) -> out[cols][rows] (bf16)
__global__ __launch_bounds__(256) void transpose_to_bf16(const float* __restrict__ in,
                                                         __bf16* __restrict__ out,
                                                         int rows, int cols) {
  int i = blockIdx.x * 256 + threadIdx.x;
  if (i < rows * cols) {
    int r = i / cols, c = i % cols;
    out[(size_t)c * rows + r] = (__bf16)in[i];
  }
}

// --------------------- K1: gates_pre = x@Wih + bh --------------------------

__global__ __launch_bounds__(512) void lstm_precompute(
    const __bf16* __restrict__ xb,     // [BT][256]
    const __bf16* __restrict__ WihT,   // [1024][256]
    const float*  __restrict__ bh,     // [1024]
    __bf16* __restrict__ gp)           // [BT][1024]
{
  const int lane = threadIdx.x & 31;
  const int wave = threadIdx.x >> 5;
  const int mt   = blockIdx.x;
  const int colL = lane & 15;
  const int hi   = lane >> 4;
  const int kb   = hi * 8;

  const __bf16* arow = xb + (size_t)(mt * 16 + colL) * D_;
  v16bf a[8];
#pragma unroll
  for (int kt = 0; kt < 8; ++kt) a[kt] = load_frag(arow + kt * 32 + kb);

#pragma unroll
  for (int j = 0; j < 4; ++j) {
    const int n = (wave + 16 * j) * 16 + colL;
    float b = bh[n];
    v8f acc;
#pragma unroll
    for (int r = 0; r < 8; ++r) acc[r] = b;

    const __bf16* wrow = WihT + (size_t)n * D_;
#pragma unroll
    for (int kt = 0; kt < 8; ++kt)
      acc = wmma_bf16(a[kt], load_frag(wrow + kt * 32 + kb), acc);

#pragma unroll
    for (int r = 0; r < 8; ++r)
      gp[(size_t)(mt * 16 + r + 8 * hi) * G4H + n] = (__bf16)acc[r];
  }
}

// --------------------- K2: sequential LSTM recurrence ----------------------
// 4 WGs x 512 threads. WG owns 16 batch rows; wave w owns H-cols [w*16,w*16+16).

__global__ __launch_bounds__(512) void lstm_recurrent(
    const __bf16* __restrict__ gp,     // [BT][1024] (x@Wih + bh)
    const __bf16* __restrict__ WhhT,   // [1024][256]
    const float*  __restrict__ c0,     // [B][256]
    const float*  __restrict__ h0,     // [B][256]
    __bf16* __restrict__ hseq)         // [B][T][256]
{
  __shared__ __bf16 h_lds[2][16 * H_];      // 16 KB: double-buffered h (bf16)
  __shared__ __bf16 w_lds[512 * H_];        // 256 KB: WhhT rows [0,512) = gates i,f

  const int lane  = threadIdx.x & 31;
  const int wave  = threadIdx.x >> 5;
  const int bbase = blockIdx.x * 16;
  const int colL  = lane & 15;
  const int hi    = lane >> 4;
  const int kb    = hi * 8;
  const int hc    = wave * 16 + colL;       // this lane's H column

  // cooperative copy: gates i,f of WhhT -> LDS (16B chunks, 32 per thread)
  {
    const v8bf* src = (const v8bf*)WhhT;
    v8bf* dst = (v8bf*)w_lds;
    for (int i = threadIdx.x; i < (512 * H_) / 8; i += 512) dst[i] = src[i];
  }

  // c tile in registers (C-layout: elem r -> row r+8*hi, col hc)
  v8f c;
#pragma unroll
  for (int r = 0; r < 8; ++r)
    c[r] = c0[(size_t)(bbase + r + 8 * hi) * H_ + hc];

  // seed buffer 0 with h0
  for (int i = threadIdx.x; i < 16 * H_; i += 512)
    h_lds[0][i] = (__bf16)h0[(size_t)(bbase + (i >> 8)) * H_ + (i & 255)];
  __syncthreads();

  // A-fragments of current h (persist across the step)
  v16bf afrag[8];
  {
    const __bf16* hrow = &h_lds[0][colL * H_];
#pragma unroll
    for (int kt = 0; kt < 8; ++kt) afrag[kt] = load_frag(hrow + kt * 32 + kb);
  }

  // per-lane bases
  const __bf16* gplane = gp + (size_t)bbase * T_ * G4H + hc;
  const __bf16* wlds0  = &w_lds[(size_t)hc * H_];                 // gate i
  const __bf16* wlds1  = &w_lds[(size_t)(256 + hc) * H_];         // gate f
  const __bf16* wg2    = WhhT + (size_t)(512 + hc) * H_;          // gate g
  const __bf16* wg3    = WhhT + (size_t)(768 + hc) * H_;          // gate o

  // gates_pre prefetch buffers (one full step ahead)
  __bf16 gpA[4][8], gpB[4][8];
#pragma unroll
  for (int g = 0; g < 4; ++g)
#pragma unroll
    for (int r = 0; r < 8; ++r)
      gpA[g][r] = gplane[((size_t)(r + 8 * hi) * T_) * G4H + g * 256];  // t = 0

  auto step = [&](int t, int rw, __bf16 gcur[4][8], __bf16 gnxt[4][8]) {
    // ---- prefetch gates_pre for t+1 (consumed next step) ----
    const int tn = t + 1;
    if (tn < T_) {
#pragma unroll
      for (int g = 0; g < 4; ++g)
#pragma unroll
        for (int r = 0; r < 8; ++r)
          gnxt[g][r] = gplane[((size_t)(r + 8 * hi) * T_ + tn) * G4H + g * 256];
    }

    // ---- gates = h @ Whh : i,f from LDS; g,o from L2 (double-buffered) ----
    v8f acc[4] = {v8f{}, v8f{}, v8f{}, v8f{}};
    v16bf bg[2][2];
    bg[0][0] = load_frag(wg2 + kb);
    bg[0][1] = load_frag(wg3 + kb);
#pragma unroll
    for (int kt = 0; kt < 8; ++kt) {
      const int cur = kt & 1;
      if (kt < 7) {                      // issue next global frags first
        bg[cur ^ 1][0] = load_frag(wg2 + (kt + 1) * 32 + kb);
        bg[cur ^ 1][1] = load_frag(wg3 + (kt + 1) * 32 + kb);
      }
      v16bf b0 = load_frag(wlds0 + kt * 32 + kb);   // ds_load_b128 x2
      v16bf b1 = load_frag(wlds1 + kt * 32 + kb);
      acc[0] = wmma_bf16(afrag[kt], b0, acc[0]);
      acc[1] = wmma_bf16(afrag[kt], b1, acc[1]);
      acc[2] = wmma_bf16(afrag[kt], bg[cur][0], acc[2]);
      acc[3] = wmma_bf16(afrag[kt], bg[cur][1], acc[3]);
    }

    // ---- LSTM elementwise in registers (uses prefetched gates_pre) ----
    v8f hnew;
#pragma unroll
    for (int r = 0; r < 8; ++r) {
      float ig = sigf(acc[0][r] + (float)gcur[0][r]);
      float fg = sigf(acc[1][r] + (float)gcur[1][r]);
      float gg = tanh_fast(acc[2][r] + (float)gcur[2][r]);
      float og = sigf(acc[3][r] + (float)gcur[3][r]);
      float cn = fg * c[r] + ig * gg;
      c[r] = cn;
      hnew[r] = og * tanh_fast(cn);
    }

    // ---- publish h_new: LDS (other buffer) + global h stream (no wait) ----
#pragma unroll
    for (int r = 0; r < 8; ++r) {
      __bf16 hb = (__bf16)hnew[r];
      h_lds[rw][(r + 8 * hi) * H_ + hc] = hb;
      hseq[((size_t)(bbase + r + 8 * hi) * T_ + t) * H_ + hc] = hb;
    }
    __syncthreads();   // single barrier/step (double-buffered h)

    // ---- reload A-fragments of h_new ----
    const __bf16* hrow = &h_lds[rw][colL * H_];
#pragma unroll
    for (int kt = 0; kt < 8; ++kt) afrag[kt] = load_frag(hrow + kt * 32 + kb);
  };

  for (int t = 0; t < T_; t += 2) {
    step(t,     1, gpA, gpB);
    step(t + 1, 0, gpB, gpA);
  }
}

// --------------------- K3: out = hseq @ Wout + bout ------------------------

__global__ __launch_bounds__(512) void lstm_project(
    const __bf16* __restrict__ hseq,   // [BT][256]
    const __bf16* __restrict__ WoutT,  // [256][256]
    const float*  __restrict__ bout,   // [256]
    float* __restrict__ out)           // [BT][256]
{
  const int lane = threadIdx.x & 31;
  const int wave = threadIdx.x >> 5;
  const int mt   = blockIdx.x;
  const int colL = lane & 15;
  const int hi   = lane >> 4;
  const int kb   = hi * 8;
  const int n    = wave * 16 + colL;

  const __bf16* arow = hseq + (size_t)(mt * 16 + colL) * H_;
  v16bf a[8];
#pragma unroll
  for (int kt = 0; kt < 8; ++kt) a[kt] = load_frag(arow + kt * 32 + kb);

  float b = bout[n];
  v8f acc;
#pragma unroll
  for (int r = 0; r < 8; ++r) acc[r] = b;

  const __bf16* wrow = WoutT + (size_t)n * H_;
#pragma unroll
  for (int kt = 0; kt < 8; ++kt)
    acc = wmma_bf16(a[kt], load_frag(wrow + kt * 32 + kb), acc);

#pragma unroll
  for (int r = 0; r < 8; ++r)
    out[(size_t)(mt * 16 + r + 8 * hi) * O_ + n] = acc[r];
}

// ---------------------------------------------------------------------------

extern "C" void kernel_launch(void* const* d_in, const int* in_sizes, int n_in,
                              void* d_out, int out_size, void* d_ws, size_t ws_size,
                              hipStream_t stream) {
  const float* x    = (const float*)d_in[0];   // [B,T,D]
  const float* c0   = (const float*)d_in[1];   // [B,H]
  const float* h0   = (const float*)d_in[2];   // [B,H]
  const float* Wih  = (const float*)d_in[3];   // [D,4H]
  const float* Whh  = (const float*)d_in[4];   // [H,4H]
  const float* bh   = (const float*)d_in[5];   // [4H]
  const float* Wout = (const float*)d_in[6];   // [H,O]
  const float* bout = (const float*)d_in[7];   // [O]
  float* out = (float*)d_out;                  // [B,T,O]

  char* ws = (char*)d_ws;
  __bf16* gpb   = (__bf16*)(ws + GP_OFF);
  __bf16* xb    = (__bf16*)(ws + XB_OFF);
  __bf16* hseq  = (__bf16*)(ws + HS_OFF);
  __bf16* WihT  = (__bf16*)(ws + WIHT_OFF);
  __bf16* WhhT  = (__bf16*)(ws + WHHT_OFF);
  __bf16* WoutT = (__bf16*)(ws + WOUTT_OFF);

  cvt_f32_to_bf16<<<(BT * D_) / 256, 256, 0, stream>>>(x, xb, BT * D_);
  transpose_to_bf16<<<(D_ * G4H) / 256, 256, 0, stream>>>(Wih,  WihT,  D_, G4H);
  transpose_to_bf16<<<(H_ * G4H) / 256, 256, 0, stream>>>(Whh,  WhhT,  H_, G4H);
  transpose_to_bf16<<<(H_ * O_)  / 256, 256, 0, stream>>>(Wout, WoutT, H_, O_);

  lstm_precompute<<<BT / 16, 512, 0, stream>>>(xb, WihT, bh, gpb);
  lstm_recurrent<<<B_ / 16, 512, 0, stream>>>(gpb, WhhT, c0, h0, hseq);
  lstm_project<<<BT / 16, 512, 0, stream>>>(hseq, WoutT, bout, out);
}
